// GCN_82111184765289
// MI455X (gfx1250) — matrix operand order repaired and checked by reading, not compiled, not run on hardware
//
#include <hip/hip_runtime.h>
#include <hip/hip_bf16.h>

typedef __bf16 bf16;
typedef __attribute__((ext_vector_type(16))) __bf16 v16bf;
typedef __attribute__((ext_vector_type(8)))  __bf16 v8bf;
typedef __attribute__((ext_vector_type(4)))  __bf16 v4bf;
typedef __attribute__((ext_vector_type(8)))  float  v8f;
typedef __attribute__((ext_vector_type(4)))  float  v4f;

#define NNODES 100000
#define NEDGES 3200000

// ---------------- utility: zero-fill (graph-capture-safe memset) ----------------
__global__ void k_zero(float* __restrict__ p, int n4) {
    int i = blockIdx.x * blockDim.x + threadIdx.x;
    if (i < n4) ((v4f*)p)[i] = (v4f){0.f, 0.f, 0.f, 0.f};
}

// ---------------- degrees via f32 atomics ----------------
__global__ void k_degree(const int* __restrict__ src, const int* __restrict__ dst,
                         float* __restrict__ dout, float* __restrict__ din, int E) {
    int e = blockIdx.x * blockDim.x + threadIdx.x;
    if (e < E) {
        atomicAdd(dout + src[e], 1.0f);
        atomicAdd(din  + dst[e], 1.0f);
    }
}

// deg -> rsqrt(max(deg,1)) in place
__global__ void k_norm(float* __restrict__ d, int n) {
    int i = blockIdx.x * blockDim.x + threadIdx.x;
    if (i < n) { float v = d[i]; d[i] = rsqrtf(v < 1.0f ? 1.0f : v); }
}

// ---------------- edge scatter: agg[dst] += h[src] * onorm[src] ----------------
// one wave per edge, each lane handles 4 consecutive features (F=128)
__global__ void k_scatter(const float* __restrict__ h, const int* __restrict__ src,
                          const int* __restrict__ dst, const float* __restrict__ onorm,
                          float* __restrict__ agg, int E) {
    int gid = blockIdx.x * blockDim.x + threadIdx.x;
    int e = gid >> 5;
    if (e >= E) return;
    int c = (gid & 31) << 2;
    int s = src[e], d = dst[e];
    float w = onorm[s];
    v4f v = *(const v4f*)(h + (size_t)s * 128 + c);
    float* o = agg + (size_t)d * 128 + c;
    atomicAdd(o + 0, v.x * w);
    atomicAdd(o + 1, v.y * w);
    atomicAdd(o + 2, v.z * w);
    atomicAdd(o + 3, v.w * w);
}

// ---------------- fp32 -> bf16 with optional per-row scale ----------------
// i indexes groups of 4 features; sh = log2(F/4)
__global__ void k_cvt(const float* __restrict__ in, const float* __restrict__ scale,
                      bf16* __restrict__ out, int sh, int total4) {
    int i = blockIdx.x * blockDim.x + threadIdx.x;
    if (i >= total4) return;
    int node = i >> sh;
    int c = (i & ((1 << sh) - 1)) << 2;
    int F = 4 << sh;
    float s = scale ? scale[node] : 1.0f;
    v4f v = *(const v4f*)(in + (size_t)node * F + c);
    v4bf o;
    o[0] = (bf16)(v.x * s); o[1] = (bf16)(v.y * s);
    o[2] = (bf16)(v.z * s); o[3] = (bf16)(v.w * s);
    *(v4bf*)(out + (size_t)node * F + c) = o;
}

// ---------------- pack fp32 weights [K][N] into WMMA-B fragment layout ----------
// Bp index = ((nt*(K/32) + kt)*32 + lane)*16 ; lane holds 16 consecutive K values
// for column n = nt*16 + (lane&15), starting at kt*32 + (lane&16 ? 16 : 0).
__global__ void k_packw(const float* __restrict__ W, bf16* __restrict__ Bp, int K, int N) {
    int t = blockIdx.x * blockDim.x + threadIdx.x;
    int total = (N >> 4) * (K >> 5) * 32;
    if (t >= total) return;
    int lane = t & 31;
    int kt = (t >> 5) % (K >> 5);
    int nt = (t >> 5) / (K >> 5);
    int n  = (nt << 4) + (lane & 15);
    int kb = (kt << 5) + ((lane & 16) ? 16 : 0);
    bf16* o = Bp + (size_t)t * 16;
    for (int j = 0; j < 16; ++j)
        o[j] = (bf16)W[(size_t)(kb + j) * N + n];
}

// ---------------- bf16 WMMA GEMM: out = act(A @ B + bias) ----------------
// A: [M x K] bf16 row-major; Bp: packed fragments; out: [M x Nout] fp32
// block = 128 threads (4 waves); each wave -> one 16x16 tile.
__global__ void k_gemm_bf16(const bf16* __restrict__ A, const bf16* __restrict__ Bp,
                            const float* __restrict__ bias, float* __restrict__ out,
                            int K, int Nout, int relu) {
    int lane = threadIdx.x & 31;
    int wave = threadIdx.x >> 5;
    int m0 = blockIdx.x << 4;
    int nt = (blockIdx.y << 2) + wave;        // global 16-col tile index
    int row = m0 + (lane & 15);
    int kb  = (lane & 16) ? 16 : 0;           // A fragment K base per half-wave
    const bf16* arow = A + (size_t)row * K;
    const bf16* bbase = Bp + ((size_t)nt * (K >> 5)) * 512;

    v8f acc = {};
    int ksteps = K >> 5;
    for (int kt = 0; kt < ksteps; ++kt) {
        v8bf alo = *(const v8bf*)(arow + (kt << 5) + kb);
        v8bf ahi = *(const v8bf*)(arow + (kt << 5) + kb + 8);
        v16bf a = __builtin_shufflevector(alo, ahi,
                    0,1,2,3,4,5,6,7,8,9,10,11,12,13,14,15);
        v16bf b = *(const v16bf*)(bbase + ((size_t)((kt << 5) + lane) << 4));
        acc = __builtin_amdgcn_wmma_f32_16x16x32_bf16(
                  false, a, false, b, (short)0, acc, false, false);
    }

    int n = (nt << 4) + (lane & 15);
    float bs = bias[n];
    int mbase = m0 + ((lane & 16) ? 8 : 0);
    for (int r = 0; r < 8; ++r) {
        float v = acc[r] + bs;
        if (relu) v = v > 0.f ? v : 0.f;
        out[(size_t)(mbase + r) * Nout + n] = v;
    }
}

// ---------------- host-side orchestration ----------------
extern "C" void kernel_launch(void* const* d_in, const int* in_sizes, int n_in,
                              void* d_out, int out_size, void* d_ws, size_t ws_size,
                              hipStream_t stream) {
    const float* x   = (const float*)d_in[0];
    const int*   src = (const int*)  d_in[1];
    const int*   dst = (const int*)  d_in[2];
    const float* w1  = (const float*)d_in[3];
    const float* b1  = (const float*)d_in[4];
    const float* w2  = (const float*)d_in[5];
    const float* b2  = (const float*)d_in[6];
    const float* wm1 = (const float*)d_in[7];
    const float* bm1 = (const float*)d_in[8];
    const float* wm2 = (const float*)d_in[9];
    const float* bm2 = (const float*)d_in[10];
    float* out = (float*)d_out;

    const int N = NNODES, E = NEDGES;

    // workspace carve-out (256B aligned)
    char* p = (char*)d_ws;
    auto carve = [&](size_t bytes) -> void* {
        void* r = (void*)p;
        p += (bytes + 255) & ~(size_t)255;
        return r;
    };
    float* onorm = (float*)carve((size_t)N * 4);            // out-degree -> norm
    float* inorm = (float*)carve((size_t)N * 4);            // in-degree  -> norm
    float* agg   = (float*)carve((size_t)N * 128 * 4);      // scatter accumulator
    bf16*  abf   = (bf16*) carve((size_t)N * 256 * 2);      // bf16 GEMM A (max K=256)
    float* h1    = (float*)carve((size_t)N * 128 * 4);      // conv1 out
    float* h2    = (float*)carve((size_t)N * 128 * 4);      // conv2 out
    float* h3    = (float*)carve((size_t)N * 256 * 4);      // mlp hidden
    bf16*  w1b   = (bf16*) carve((size_t)128 * 128 * 2);
    bf16*  w2b   = (bf16*) carve((size_t)128 * 128 * 2);
    bf16*  wm1b  = (bf16*) carve((size_t)128 * 256 * 2);
    bf16*  wm2b  = (bf16*) carve((size_t)256 * 64  * 2);

    const int TB = 256;
    auto cdiv = [](int a, int b) { return (a + b - 1) / b; };

    // degrees -> symmetric norms
    k_zero<<<cdiv(N / 4, TB), TB, 0, stream>>>(onorm, N / 4);
    k_zero<<<cdiv(N / 4, TB), TB, 0, stream>>>(inorm, N / 4);
    k_degree<<<cdiv(E, TB), TB, 0, stream>>>(src, dst, onorm, inorm, E);
    k_norm<<<cdiv(N, TB), TB, 0, stream>>>(onorm, N);
    k_norm<<<cdiv(N, TB), TB, 0, stream>>>(inorm, N);

    // pack weights to bf16 WMMA-B fragment layout
    k_packw<<<cdiv((128/16)*(128/32)*32, TB), TB, 0, stream>>>(w1,  w1b,  128, 128);
    k_packw<<<cdiv((128/16)*(128/32)*32, TB), TB, 0, stream>>>(w2,  w2b,  128, 128);
    k_packw<<<cdiv((256/16)*(128/32)*32, TB), TB, 0, stream>>>(wm1, wm1b, 128, 256);
    k_packw<<<cdiv((64/16)*(256/32)*32,  TB), TB, 0, stream>>>(wm2, wm2b, 256, 64);

    const int scat_blocks = cdiv(E * 32, TB);     // one wave per edge
    const int M16 = N / 16;                       // 6250, exact

    // ---- conv1: agg = scatter(x * onorm) ; h1 = relu((agg*inorm) @ w1 + b1)
    k_zero<<<cdiv(N * 128 / 4, TB), TB, 0, stream>>>(agg, N * 128 / 4);
    k_scatter<<<scat_blocks, TB, 0, stream>>>(x, src, dst, onorm, agg, E);
    k_cvt<<<cdiv(N * 32, TB), TB, 0, stream>>>(agg, inorm, abf, 5, N * 32);
    k_gemm_bf16<<<dim3(M16, 128 / 64), 128, 0, stream>>>(abf, w1b, b1, h1, 128, 128, 1);

    // ---- conv2
    k_zero<<<cdiv(N * 128 / 4, TB), TB, 0, stream>>>(agg, N * 128 / 4);
    k_scatter<<<scat_blocks, TB, 0, stream>>>(h1, src, dst, onorm, agg, E);
    k_cvt<<<cdiv(N * 32, TB), TB, 0, stream>>>(agg, inorm, abf, 5, N * 32);
    k_gemm_bf16<<<dim3(M16, 128 / 64), 128, 0, stream>>>(abf, w2b, b2, h2, 128, 128, 1);

    // ---- MLP hidden: h3 = relu(h2 @ wm1 + bm1)   [128 -> 256]
    k_cvt<<<cdiv(N * 32, TB), TB, 0, stream>>>(h2, (const float*)nullptr, abf, 5, N * 32);
    k_gemm_bf16<<<dim3(M16, 256 / 64), 128, 0, stream>>>(abf, wm1b, bm1, h3, 128, 256, 1);

    // ---- MLP out: out = h3 @ wm2 + bm2           [256 -> 64], no relu
    k_cvt<<<cdiv(N * 64, TB), TB, 0, stream>>>(h3, (const float*)nullptr, abf, 6, N * 64);
    k_gemm_bf16<<<dim3(M16, 64 / 64), 128, 0, stream>>>(abf, wm2b, bm2, out, 256, 64, 0);
}
